// MentionEncoder_20246475834104
// MI455X (gfx1250) — matrix operand order repaired and checked by reading, not compiled, not run on hardware
//
#include <hip/hip_runtime.h>
#include <hip/hip_bf16.h>
#include <math.h>

// ---------------------------------------------------------------------------
// Types for CDNA5 WMMA (wave32, 16x16x32 bf16 -> f32)
// ---------------------------------------------------------------------------
typedef __attribute__((ext_vector_type(16))) __bf16 v16bf;
typedef __attribute__((ext_vector_type(8)))  __bf16 v8bf;
typedef __attribute__((ext_vector_type(8)))  float  v8f;

#define TILE_M 128
#define TILE_N 128
#define TILE_K 32
#define LDS_STRIDE 40   // halfs; 80 bytes = multiple of 16B -> aligned vector loads

// Pack two floats into two bf16 halves of one dword (RNE via compiler fptrunc;
// clang can fuse the pair into a packed convert where the ISA provides one).
__device__ __forceinline__ unsigned pk2bf(float a, float b) {
  unsigned short lo = __builtin_bit_cast(unsigned short, (__bf16)a);
  unsigned short hi = __builtin_bit_cast(unsigned short, (__bf16)b);
  return ((unsigned)hi << 16) | (unsigned)lo;
}
__device__ __forceinline__ float gelu_exact(float x) {
  return 0.5f * x * (1.0f + erff(x * 0.70710678118654752f));
}
// Two contiguous 16B LDS loads -> one v16bf fragment (no element shuffling).
__device__ __forceinline__ v16bf frag2x8(const unsigned short* p0,
                                         const unsigned short* p1) {
  v8bf lo = *(const v8bf*)p0;
  v8bf hi = *(const v8bf*)p1;
  return __builtin_shufflevector(lo, hi, 0, 1, 2, 3, 4, 5, 6, 7,
                                 8, 9, 10, 11, 12, 13, 14, 15);
}

// ---------------------------------------------------------------------------
// scores[i] = b_s[0]   (accumulated into by the GEMM epilogue)
// ---------------------------------------------------------------------------
__global__ void init_scores_kernel(float* __restrict__ scores,
                                   const float* __restrict__ b_s, int total) {
  int i = blockIdx.x * blockDim.x + threadIdx.x;
  if (i < total) scores[i] = b_s[0];
}

// ---------------------------------------------------------------------------
// Fused span-feature gather + GEMM (bf16 WMMA) + bias + GELU + score partials.
// Block = 256 threads (8 wave32), computes a 128(M) x 128(N) tile.
// Wave grid 4x2; each wave owns 32x64 = 2x4 WMMA 16x16 tiles (8 WMMA/iter).
// ---------------------------------------------------------------------------
__global__ __launch_bounds__(256)
void span_gemm_kernel(const float* __restrict__ emb,   // (B,N,H)
                      const int*   __restrict__ spans, // (S,2)
                      const float* __restrict__ wemb,  // (MAX_W, width_d)
                      const float* __restrict__ W1,    // (span_dim, H)
                      const float* __restrict__ b1,    // (H)
                      const float* __restrict__ w_s,   // (H)
                      float* __restrict__ h_ws,        // (B,S,H)
                      float* __restrict__ scores,      // (B,S)
                      int S, int N, int H, int span_dim, int width_d)
{
  // A: row-major [m][k]; B: TRANSPOSED [n][k] so fragment K-runs are contiguous.
  __shared__ __align__(16) unsigned short As[TILE_M][LDS_STRIDE];
  __shared__ __align__(16) unsigned short Bt[TILE_N][LDS_STRIDE];
  __shared__ int sh_start[TILE_M], sh_end[TILE_M];

  const int tid    = threadIdx.x;
  const int lane   = tid & 31;
  const int wave   = tid >> 5;
  const int b      = blockIdx.z;
  const int m_base = blockIdx.x * TILE_M;
  const int n0     = blockIdx.y * TILE_N;

  if (tid < TILE_M) {
    int sp = m_base + tid;
    if (sp >= S) sp = S - 1;             // clamp; OOB rows masked at store time
    sh_start[tid] = spans[2 * sp];
    sh_end[tid]   = spans[2 * sp + 1];
  }
  __syncthreads();

  const int wm = (wave >> 1) * 32;       // wave M offset within tile
  const int wn = (wave & 1) * 64;        // wave N offset within tile
  v8f acc[2][4] = {};

  const size_t embB = (size_t)b * N * H;

  // A-fill: thread handles 2 consecutive k (float2 load -> packed b32 store).
  // Half-wave (16 lanes) per row; lanes cover the 32-wide K slab.
  const int a_m0 = tid >> 4;             // 0..15, +it*16
  const int a_k2 = (tid & 15) * 2;       // even k within slab
  // B-fill: contiguous n across threads -> coalesced W1 reads; packed k-pair
  // stored as one b32 into the transposed LDS tile.
  const int b_nn = tid & 127;
  const int b_k2 = (tid >> 7) * 2;       // 0 or 2, +it*4

  for (int k0 = 0; k0 < span_dim; k0 += TILE_K) {
    // Region of this K-slab is UNIFORM when the slab fits in one segment
    // (true for every slab when H % TILE_K == 0): scalar branch, no exec masks.
    int rgn;
    if (k0 + TILE_K <= H)                              rgn = 0; // x
    else if (k0 >= H && k0 + TILE_K <= 2 * H)          rgn = 1; // y
    else if (k0 >= 2 * H && k0 + TILE_K <= 3 * H)      rgn = 2; // x*y
    else                                               rgn = 3; // generic

    if (rgn == 0) {
      #pragma unroll
      for (int it = 0; it < 8; ++it) {
        int m = a_m0 + it * 16;
        const float2 v = *(const float2*)&emb[embB + (size_t)sh_start[m] * H +
                                              (k0 + a_k2)];
        *(unsigned*)&As[m][a_k2] = pk2bf(v.x, v.y);
      }
    } else if (rgn == 1) {
      #pragma unroll
      for (int it = 0; it < 8; ++it) {
        int m = a_m0 + it * 16;
        const float2 v = *(const float2*)&emb[embB + (size_t)sh_end[m] * H +
                                              (k0 - H + a_k2)];
        *(unsigned*)&As[m][a_k2] = pk2bf(v.x, v.y);
      }
    } else if (rgn == 2) {
      #pragma unroll
      for (int it = 0; it < 8; ++it) {
        int m  = a_m0 + it * 16;
        int kq = k0 - 2 * H + a_k2;
        const float2 vx = *(const float2*)&emb[embB + (size_t)sh_start[m] * H + kq];
        const float2 vy = *(const float2*)&emb[embB + (size_t)sh_end[m] * H + kq];
        *(unsigned*)&As[m][a_k2] = pk2bf(vx.x * vy.x, vx.y * vy.y);
      }
    } else {
      // Boundary slab(s) only: per-element generic gather.
      #pragma unroll
      for (int it = 0; it < 8; ++it) {
        int m = a_m0 + it * 16;
        int st = sh_start[m], en = sh_end[m];
        float v2[2];
        #pragma unroll
        for (int j = 0; j < 2; ++j) {
          int k = k0 + a_k2 + j;
          float v = 0.0f;
          if (k < H)            v = emb[embB + (size_t)st * H + k];
          else if (k < 2 * H)   v = emb[embB + (size_t)en * H + (k - H)];
          else if (k < 3 * H) {
            int kq = k - 2 * H;
            v = emb[embB + (size_t)st * H + kq] *
                emb[embB + (size_t)en * H + kq];
          } else if (k < span_dim) {
            v = wemb[(en - st) * width_d + (k - 3 * H)];
          }
          v2[j] = v;
        }
        *(unsigned*)&As[m][a_k2] = pk2bf(v2[0], v2[1]);
      }
    }

    // B tile (W1[k0:k0+32, n0:n0+128]) -> transposed bf16 LDS Bt[n][k]
    #pragma unroll
    for (int it = 0; it < 8; ++it) {
      int kk = b_k2 + it * 4;
      int k  = k0 + kk;
      int n  = n0 + b_nn;
      float v0 = (k < span_dim && n < H) ? W1[(size_t)k * H + n] : 0.0f;
      float v1 = (k + 1 < span_dim && n < H) ? W1[(size_t)(k + 1) * H + n] : 0.0f;
      *(unsigned*)&Bt[b_nn][kk] = pk2bf(v0, v1);
    }
    if (tid == 0 && (k0 + TILE_K) < span_dim) {
      __builtin_prefetch(&W1[(size_t)(k0 + TILE_K) * H + n0], 0, 1);
    }
    __syncthreads();

    // ---- Fragments per ISA 7.12.2 layouts (all 16B-aligned ds_load_b128).
    // A 16x32: lane -> M = lane&15; khalf = (lane>>4)*8;
    //          elements 0..7 = K[khalf..+7], 8..15 = K[khalf+16..+23].
    const int khalf = (lane >> 4) * 8;
    v16bf afrag[2], bfrag[4];
    #pragma unroll
    for (int mi = 0; mi < 2; ++mi) {
      const unsigned short* row = &As[wm + mi * 16 + (lane & 15)][0];
      afrag[mi] = frag2x8(row + khalf, row + khalf + 16);
    }
    // B 32x16: lane -> N = lane&15; same K order as A (khalf..+7, khalf+16..+23).
    const int kbase = khalf;
    #pragma unroll
    for (int ni = 0; ni < 4; ++ni) {
      const unsigned short* rowb = &Bt[wn + ni * 16 + (lane & 15)][0];
      bfrag[ni] = frag2x8(rowb + kbase, rowb + kbase + 16);
    }
    #pragma unroll
    for (int mi = 0; mi < 2; ++mi)
      #pragma unroll
      for (int ni = 0; ni < 4; ++ni)
        acc[mi][ni] = __builtin_amdgcn_wmma_f32_16x16x32_bf16(
            false, afrag[mi], false, bfrag[ni], (short)0, acc[mi][ni],
            false, false);
    __syncthreads();
  }

  // ---- Epilogue: +b1, exact GELU, store h, partial score dot w_s.
  // C/D layout: VGPR r, lane l -> (M = r + 8*(l>>4), N = l&15).
  #pragma unroll
  for (int mi = 0; mi < 2; ++mi) {
    #pragma unroll
    for (int r = 0; r < 8; ++r) {
      int row  = wm + mi * 16 + r + ((lane >> 4) * 8);
      int span = m_base + row;
      bool ok  = span < S;
      float partial = 0.0f;
      #pragma unroll
      for (int ni = 0; ni < 4; ++ni) {
        int n = n0 + wn + ni * 16 + (lane & 15);
        if (n < H) {
          float x  = acc[mi][ni][r] + b1[n];
          float hv = gelu_exact(x);
          if (ok) {
            h_ws[((size_t)b * S + span) * H + n] = hv;
            partial += hv * w_s[n];
          }
        }
      }
      // 16-lane groups share one row: butterfly-reduce within group.
      partial += __shfl_xor(partial, 1, 32);
      partial += __shfl_xor(partial, 2, 32);
      partial += __shfl_xor(partial, 4, 32);
      partial += __shfl_xor(partial, 8, 32);
      if (ok && (lane & 15) == 0)
        atomicAdd(&scores[(size_t)b * S + span], partial);
    }
  }
}

// ---------------------------------------------------------------------------
// Deterministic top-K: MSB-first radix select + in-index-order selection scan.
// ---------------------------------------------------------------------------
__device__ __forceinline__ unsigned map_f32(float f) {
  unsigned u = __builtin_bit_cast(unsigned, f);
  return (u & 0x80000000u) ? ~u : (u | 0x80000000u);
}

__device__ int excl_scan256(int v, int* sh) {
  int tid = threadIdx.x;
  sh[tid] = v;
  __syncthreads();
  #pragma unroll
  for (int off = 1; off < 256; off <<= 1) {
    int t = (tid >= off) ? sh[tid - off] : 0;
    __syncthreads();
    sh[tid] += t;
    __syncthreads();
  }
  int incl = sh[tid];
  __syncthreads();
  return incl - v;
}

__global__ __launch_bounds__(256)
void topk_kernel(const float* __restrict__ scores, int S, int K,
                 int* __restrict__ sel)
{
  const int tid = threadIdx.x;
  const int b   = blockIdx.x;
  const float* sc = scores + (size_t)b * S;

  __shared__ unsigned hist[256];
  __shared__ unsigned sh_prefix;
  __shared__ int sh_rem;
  __shared__ int sh_scan[256];
  __shared__ int out_run, tie_run;

  unsigned prefix = 0, donemask = 0;
  int remaining = K;
  for (int pass = 3; pass >= 0; --pass) {
    int shift = pass * 8;
    hist[tid] = 0;
    __syncthreads();
    for (int i = tid; i < S; i += 256) {
      unsigned u = map_f32(sc[i]);
      if ((u & donemask) == prefix)
        atomicAdd(&hist[(u >> shift) & 255u], 1u);
    }
    __syncthreads();
    if (tid == 0) {
      int rem = remaining;
      unsigned d = 0;
      for (int j = 255; j >= 0; --j) {
        int c = (int)hist[j];
        if (c >= rem) { d = (unsigned)j; break; }
        rem -= c;
      }
      sh_prefix = prefix | (d << shift);
      sh_rem    = rem;
    }
    __syncthreads();
    prefix    = sh_prefix;
    remaining = sh_rem;
    donemask |= (255u << shift);
    __syncthreads();
  }
  const unsigned T = prefix;          // exact key of the K-th largest element
  const int ties_needed = remaining;  // how many ==T elements to take (by index)
  if (tid == 0) { out_run = 0; tie_run = 0; }
  __syncthreads();

  for (int base = 0; base < S; base += 256) {
    int i = base + tid;
    unsigned u = (i < S) ? map_f32(sc[i]) : 0u;
    int gt  = (i < S) && (u > T);
    int tie = (i < S) && (u == T);
    int tie_ex = excl_scan256(tie, sh_scan);
    int take = gt || (tie && (tie_run + tie_ex) < ties_needed);
    int sel_ex = excl_scan256(take, sh_scan);
    int pos = out_run + sel_ex;       // index-ascending by construction
    if (take && pos < K) sel[(size_t)b * K + pos] = i;
    __syncthreads();
    if (tid == 255) {
      tie_run += tie_ex + tie;
      out_run += sel_ex + take;
    }
    __syncthreads();
  }
}

// ---------------------------------------------------------------------------
// Gather selected rows into d_out: embs (B,K,H), scores (B,K), spans (B,K,2)
// ---------------------------------------------------------------------------
__global__ void gather_kernel(const float* __restrict__ h_ws,
                              const float* __restrict__ scores,
                              const int* __restrict__ sel,
                              const int* __restrict__ spans,
                              float* __restrict__ out_embs,
                              float* __restrict__ out_scores,
                              int* __restrict__ out_spans,
                              int S, int K, int H)
{
  int b = blockIdx.y, p = blockIdx.x;
  int span = sel[(size_t)b * K + p];
  const float* hr = h_ws + ((size_t)b * S + span) * H;
  float* orow = out_embs + ((size_t)b * K + p) * H;
  for (int t = threadIdx.x; t < H; t += blockDim.x) orow[t] = hr[t];
  if (threadIdx.x == 0) {
    out_scores[(size_t)b * K + p] = scores[(size_t)b * S + span];
    out_spans[((size_t)b * K + p) * 2 + 0] = spans[span * 2 + 0];
    out_spans[((size_t)b * K + p) * 2 + 1] = spans[span * 2 + 1];
  }
}

static inline size_t align256(size_t x) { return (x + 255u) & ~(size_t)255u; }

extern "C" void kernel_launch(void* const* d_in, const int* in_sizes, int n_in,
                              void* d_out, int out_size, void* d_ws, size_t ws_size,
                              hipStream_t stream) {
  const float* emb   = (const float*)d_in[0];
  const int*   spans = (const int*)  d_in[2];
  const float* wemb  = (const float*)d_in[3];
  const float* W1    = (const float*)d_in[4];
  const float* b1    = (const float*)d_in[5];
  const float* w_s   = (const float*)d_in[6];
  const float* b_s   = (const float*)d_in[7];
  (void)d_in[1]; (void)n_in; (void)out_size; (void)ws_size;

  const int H        = in_sizes[5];            // 768
  const int N        = 1024;                   // reference constant
  const int B        = in_sizes[1] / N;        // 2
  const int S        = in_sizes[2] / 2;        // 30285
  const int span_dim = in_sizes[4] / H;        // 2324
  const int width_d  = span_dim - 3 * H;       // 20
  const int K        = (N * 2) / 5;            // int(0.4*N) = 409

  // Workspace: scores (B*S f32) | sel (B*K i32) | h (B*S*H f32)
  char* ws = (char*)d_ws;
  float* scores = (float*)ws;
  size_t off = align256((size_t)B * S * sizeof(float));
  int* sel = (int*)(ws + off);
  off += align256((size_t)B * K * sizeof(int));
  float* h_ws = (float*)(ws + off);

  const int totS = B * S;
  init_scores_kernel<<<(totS + 255) / 256, 256, 0, stream>>>(scores, b_s, totS);

  dim3 grid((S + TILE_M - 1) / TILE_M, (H + TILE_N - 1) / TILE_N, B);
  span_gemm_kernel<<<grid, 256, 0, stream>>>(emb, spans, wemb, W1, b1, w_s,
                                             h_ws, scores, S, N, H, span_dim,
                                             width_d);

  topk_kernel<<<B, 256, 0, stream>>>(scores, S, K, sel);

  float* out_embs   = (float*)d_out;
  float* out_scores = out_embs + (size_t)B * K * H;
  int*   out_spans  = (int*)(out_scores + (size_t)B * K);
  dim3 ggrid(K, B);
  gather_kernel<<<ggrid, 256, 0, stream>>>(h_ws, scores, sel, spans, out_embs,
                                           out_scores, out_spans, S, K, H);
}
